// TextRegionAttentionFusion_20942260535545
// MI455X (gfx1250) — compile-verified, matrix operand order checked by reading, not run on hardware
//
#include <hip/hip_runtime.h>
#include <hip/hip_bf16.h>

// Problem constants from the reference.
#define Bn  32
#define Ln  1024
#define Hn  768
#define AHn 1024
#define Rn  64
#define KTS (Hn / 32)   // 24 K-steps of 32 (bf16 WMMA K)
#define NTS (AHn / 16)  // 64 N-tiles of 16
#define MROWS 32        // rows (M) per block: two 16-row WMMA tiles

typedef __attribute__((ext_vector_type(16))) __bf16 v16bf;
typedef __attribute__((ext_vector_type(8)))  float  v8f;

// ---------------------------------------------------------------------------
// Kernel 0: convert W2 (fp32, [H][AH] row-major) into bf16, pre-swizzled into
// the CDNA5 16-bit B-matrix (32x16) per-lane fragment layout so the GEMM can
// load each lane's 16 bf16 values as one contiguous 32-byte read.
// Fragment index f = kt*NTS + nt. Element o = (f*32 + lane)*16 + j.
// lane: n = nt*16 + (lane&15), kbase = (lane&16)?8:0
// j:    k = kt*32 + kbase + (j<8 ? j : j+8)
// ---------------------------------------------------------------------------
__global__ void pack_w2_kernel(const float* __restrict__ W2, __bf16* __restrict__ w2p) {
  int o = blockIdx.x * blockDim.x + threadIdx.x;
  if (o >= Hn * AHn) return;
  int j    = o & 15;
  int lane = (o >> 4) & 31;
  int f    = o >> 9;
  int nt   = f & (NTS - 1);
  int kt   = f >> 6;
  int n    = nt * 16 + (lane & 15);
  int kb   = (lane & 16) ? 8 : 0;
  int k    = kt * 32 + kb + (j < 8 ? j : j + 8);
  w2p[o] = (__bf16)W2[k * AHn + n];
}

// ---------------------------------------------------------------------------
// Kernel 1: alpha[row] = W1 . tanh(W2^T x_row + b2) + b1 via bf16 WMMA.
// One block = 32 rows of x (two 16-row M-tiles, staged to LDS as bf16).
// 8 waves; wave w owns n-tiles w*8 .. w*8+7 (128 columns of AH).
// Each B fragment is reused for both M-tiles: 16 WMMAs per kt-step vs
// 4 ds_load_b128 + 16 global_load_b128  -> ~1.25 loads per WMMA.
// ---------------------------------------------------------------------------
__global__ void __launch_bounds__(256)
alpha_kernel(const float* __restrict__ x, const __bf16* __restrict__ w2p,
             const float* __restrict__ b2, const float* __restrict__ W1,
             const float* __restrict__ b1, float* __restrict__ alpha) {
  __shared__ __bf16 sxa[MROWS * Hn];   // 48 KB: 32 rows x 768 bf16
  __shared__ float  salpha[MROWS];

  const int tid     = threadIdx.x;
  const int lane    = tid & 31;
  const int wave    = tid >> 5;
  const int rowbase = blockIdx.x * MROWS;

  // Stage + convert the 32-row x tile (coalesced fp32 reads).
  for (int idx = tid; idx < MROWS * Hn; idx += 256)
    sxa[idx] = (__bf16)x[(size_t)rowbase * Hn + idx];
  if (tid < MROWS) salpha[tid] = 0.0f;
  __syncthreads();

  const int m  = lane & 15;
  const int kb = (lane & 16) ? 8 : 0;

  v8f c[2][8];
#pragma unroll
  for (int t = 0; t < 2; ++t)
#pragma unroll
    for (int jj = 0; jj < 8; ++jj)
      c[t][jj] = (v8f){0.f,0.f,0.f,0.f,0.f,0.f,0.f,0.f};

  for (int kt = 0; kt < KTS; ++kt) {
    // Two A fragments (16-bit A 16x32 layout): 2 contiguous 16B LDS chunks each.
    union { uint4 q[2]; v16bf v; } af[2];
#pragma unroll
    for (int t = 0; t < 2; ++t) {
      const int abase = (t * 16 + m) * Hn + kt * 32 + kb;
      af[t].q[0] = *reinterpret_cast<const uint4*>(sxa + abase);
      af[t].q[1] = *reinterpret_cast<const uint4*>(sxa + abase + 16);
    }
#pragma unroll
    for (int jj = 0; jj < 8; ++jj) {
      const int ntg = wave * 8 + jj;
      const v16bf bf = *reinterpret_cast<const v16bf*>(
          w2p + ((size_t)(kt * NTS + ntg) * 32 + lane) * 16);
      c[0][jj] = __builtin_amdgcn_wmma_f32_16x16x32_bf16(
          false, af[0].v, false, bf, (short)0, c[0][jj], false, false);
      c[1][jj] = __builtin_amdgcn_wmma_f32_16x16x32_bf16(
          false, af[1].v, false, bf, (short)0, c[1][jj], false, false);
    }
  }

  // Epilogue: +b2, tanh, weight by W1, reduce columns.
  // C layout: lane l, vgpr i -> row (t*16 + i + (l>=16?8:0)), col (l&15) in tile.
  float acc[2][8];
#pragma unroll
  for (int t = 0; t < 2; ++t)
#pragma unroll
    for (int i = 0; i < 8; ++i) acc[t][i] = 0.0f;

#pragma unroll
  for (int jj = 0; jj < 8; ++jj) {
    const int  ncol = (wave * 8 + jj) * 16 + (lane & 15);
    const float b2v = b2[ncol];
    const float w1v = W1[ncol];
#pragma unroll
    for (int t = 0; t < 2; ++t)
#pragma unroll
      for (int i = 0; i < 8; ++i)
        acc[t][i] += tanhf(c[t][jj][i] + b2v) * w1v;
  }
  // Reduce over the 16 lanes of each half-wave (xor 8,4,2,1 stays in-half).
#pragma unroll
  for (int off = 8; off >= 1; off >>= 1)
#pragma unroll
    for (int t = 0; t < 2; ++t)
#pragma unroll
      for (int i = 0; i < 8; ++i)
        acc[t][i] += __shfl_xor(acc[t][i], off, 32);

  if ((lane & 15) == 0) {
    const int mb = (lane & 16) ? 8 : 0;
#pragma unroll
    for (int t = 0; t < 2; ++t)
#pragma unroll
      for (int i = 0; i < 8; ++i)
        atomicAdd(&salpha[t * 16 + mb + i], acc[t][i]);   // ds_add_f32
  }
  __syncthreads();
  if (tid < MROWS)
    alpha[rowbase + tid] = salpha[tid] + b1[0];
}

// ---------------------------------------------------------------------------
// Kernel 2: per-(batch,region) stable softmax over valid lines + weighted
// feature sum. region_ids are sorted per row -> binary search the range.
// ---------------------------------------------------------------------------
__global__ void __launch_bounds__(256)
fuse_kernel(const float* __restrict__ x, const int* __restrict__ region_ids,
            const int* __restrict__ line_mask, const float* __restrict__ alpha,
            float* __restrict__ out_feat, float* __restrict__ out_mask) {
  const int b   = blockIdx.x / Rn;
  const int r   = blockIdx.x % Rn;
  const int tid = threadIdx.x;

  __shared__ int   sse[2];
  __shared__ float sred[256];
  __shared__ float sw[Ln];

  if (tid == 0) {
    const int* row = region_ids + b * Ln;
    int lo = 0, hi = Ln;
    while (lo < hi) { int mid = (lo + hi) >> 1; if (row[mid] < r) lo = mid + 1; else hi = mid; }
    sse[0] = lo;
    int lo2 = lo, hi2 = Ln;
    while (lo2 < hi2) { int mid = (lo2 + hi2) >> 1; if (row[mid] < r + 1) lo2 = mid + 1; else hi2 = mid; }
    sse[1] = lo2;
  }
  __syncthreads();
  const int start = sse[0], end = sse[1], len = end - start;

  // Max over valid lines.
  float lm = -INFINITY;
  for (int i = tid; i < len; i += 256) {
    const int l = b * Ln + start + i;
    if (line_mask[l]) lm = fmaxf(lm, alpha[l]);
  }
  sred[tid] = lm; __syncthreads();
  for (int s = 128; s > 0; s >>= 1) {
    if (tid < s) sred[tid] = fmaxf(sred[tid], sred[tid + s]);
    __syncthreads();
  }
  const float mmax = sred[0];
  __syncthreads();

  // exp + denominator; cache per-line numerators in LDS.
  float ls = 0.0f;
  for (int i = tid; i < len; i += 256) {
    const int l = b * Ln + start + i;
    float e = 0.0f;
    if (line_mask[l]) e = __expf(alpha[l] - mmax);
    sw[i] = e; ls += e;
  }
  sred[tid] = ls; __syncthreads();
  for (int s = 128; s > 0; s >>= 1) {
    if (tid < s) sred[tid] += sred[tid + s];
    __syncthreads();
  }
  const float inv = 1.0f / fmaxf(sred[0], 1e-9f);
  __syncthreads();

  // Weighted feature sum (coalesced over H).
  for (int h = tid; h < Hn; h += 256) {
    float a0 = 0.0f;
    for (int i = 0; i < len; ++i)
      a0 += sw[i] * x[(size_t)(b * Ln + start + i) * Hn + h];
    out_feat[(size_t)(b * Rn + r) * Hn + h] = a0 * inv;
  }
  if (tid == 0)
    out_mask[b * Rn + r] = (mmax != -INFINITY) ? 1.0f : 0.0f;
}

// ---------------------------------------------------------------------------
extern "C" void kernel_launch(void* const* d_in, const int* in_sizes, int n_in,
                              void* d_out, int out_size, void* d_ws, size_t ws_size,
                              hipStream_t stream) {
  (void)in_sizes; (void)n_in; (void)out_size; (void)ws_size;
  const float* x          = (const float*)d_in[0];
  const int*   region_ids = (const int*)d_in[1];
  const int*   line_mask  = (const int*)d_in[2];
  const float* W2         = (const float*)d_in[3];
  const float* b2         = (const float*)d_in[4];
  const float* W1         = (const float*)d_in[5];
  const float* b1         = (const float*)d_in[6];

  float*  alpha = (float*)d_ws;                                   // 128 KB
  __bf16* w2p   = (__bf16*)((char*)d_ws + (size_t)Bn * Ln * 4);   // 1.5 MB

  float* out_feat = (float*)d_out;
  float* out_mask = out_feat + (size_t)Bn * Rn * Hn;

  pack_w2_kernel<<<(Hn * AHn + 255) / 256, 256, 0, stream>>>(W2, w2p);
  alpha_kernel<<<(Bn * Ln) / MROWS, 256, 0, stream>>>(x, w2p, b2, W1, b1, alpha);
  fuse_kernel<<<Bn * Rn, 256, 0, stream>>>(x, region_ids, line_mask, alpha,
                                           out_feat, out_mask);
}
